// DecoderLSTM_77610059039292
// MI455X (gfx1250) — matrix operand order repaired and checked by reading, not compiled, not run on hardware
//
#include <hip/hip_runtime.h>
#include <hip/hip_bf16.h>

// Problem sizes (fixed by the reference)
#define HB   32      // batch B
#define SS   512     // sequence length S
#define EE   512     // embedding dim E
#define HH   512     // hidden dim H
#define G4   2048    // 4*H gate width
#define RWG  32      // persistent-kernel workgroups (one per 16 h-columns)

typedef __attribute__((ext_vector_type(16))) __bf16 v16bf;
typedef __attribute__((ext_vector_type(8)))  __bf16 v8bf;
typedef __attribute__((ext_vector_type(8)))  float  v8f;

// ---------------------------------------------------------------------------
// Kernel 1: split W_ih (2048 x 1024 f32) into We (2048 x 512 bf16) and
//           Wh (2048 x 512 bf16), row-major [g][k]. 8 elements per thread.
// ---------------------------------------------------------------------------
__global__ void __launch_bounds__(256) k_convert_w(const float* __restrict__ W_ih,
                                                   __bf16* __restrict__ We,
                                                   __bf16* __restrict__ Wh) {
  int idx = blockIdx.x * 256 + threadIdx.x;        // over 2048*1024/8
  int g = idx >> 7;
  int e = (idx & 127) * 8;                          // 8-chunks never straddle 512
  const float4* src = (const float4*)(W_ih + (long)g * (EE + HH) + e);
  float4 x = src[0], y = src[1];
  v8bf o = { (__bf16)x.x, (__bf16)x.y, (__bf16)x.z, (__bf16)x.w,
             (__bf16)y.x, (__bf16)y.y, (__bf16)y.z, (__bf16)y.w };
  if (e < EE) *(v8bf*)(We + (long)g * EE + e)        = o;
  else        *(v8bf*)(Wh + (long)g * HH + (e - EE)) = o;
}

// ---------------------------------------------------------------------------
// Kernel 2: gather embeddings -> bf16 A matrix, rows m = s*32 + b.
//           8 elements per thread (2x float4 load, one v8bf store).
// ---------------------------------------------------------------------------
__global__ void __launch_bounds__(256) k_embed(const int* __restrict__ seq,
                                               const float* __restrict__ emb,
                                               __bf16* __restrict__ Aemb) {
  int idx = blockIdx.x * 256 + threadIdx.x;        // over 16384*512/8
  int m = idx >> 6;
  int e = (idx & 63) * 8;
  int s = m >> 5;
  int b = m & 31;
  int tok = seq[b * SS + s];
  const float4* src = (const float4*)(emb + (long)tok * EE + e);
  float4 x = src[0], y = src[1];
  v8bf o = { (__bf16)x.x, (__bf16)x.y, (__bf16)x.z, (__bf16)x.w,
             (__bf16)y.x, (__bf16)y.y, (__bf16)y.z, (__bf16)y.w };
  *(v8bf*)(Aemb + (long)m * EE + e) = o;
}

// ---------------------------------------------------------------------------
// Kernel 3: const[b][g] = h0[b] . W_hh[g] + b_ih[g] + b_hh[g]   (f32, tiny)
// ---------------------------------------------------------------------------
__global__ void __launch_bounds__(256) k_const(const float* __restrict__ h0,
                                               const float* __restrict__ W_hh,
                                               const float* __restrict__ b_ih,
                                               const float* __restrict__ b_hh,
                                               float* __restrict__ constv) {
  int idx = blockIdx.x * 256 + threadIdx.x;        // over 32*2048
  int b = idx >> 11;
  int g = idx & 2047;
  float acc = b_ih[g] + b_hh[g];
  const float* hr = h0 + b * HH;
  const float* wr = W_hh + g * HH;
  #pragma unroll 4
  for (int k = 0; k < HH; ++k) acc += hr[k] * wr[k];
  constv[idx] = acc;
}

// ---------------------------------------------------------------------------
// Kernel 4: big WMMA GEMM. Gp[m][g] = Aemb[m][:] . We[g][:] + const[m&31][g]
//           M=16384, N=2048, K=512, bf16 in, f32 accumulate.
// ---------------------------------------------------------------------------
__global__ void __launch_bounds__(256) k_gemm_emb(const __bf16* __restrict__ A,
                                                  const __bf16* __restrict__ Bw,
                                                  const float* __restrict__ constv,
                                                  float* __restrict__ Gp) {
  const int lane = threadIdx.x & 31;
  const int wave = threadIdx.x >> 5;
  const int ln   = lane & 15;
  const int half = lane >> 4;
  const int m0   = (blockIdx.y * 8 + wave) * 16;
  const int g0   = blockIdx.x * 64;

  v8f acc0 = {}, acc1 = {}, acc2 = {}, acc3 = {};
  const __bf16* arow = A + (long)(m0 + ln) * EE;

  for (int k0 = 0; k0 < EE; k0 += 32) {
    v8bf alo = *(const v8bf*)(arow + k0 + 16 * half);
    v8bf ahi = *(const v8bf*)(arow + k0 + 8 + 16 * half);
    v16bf a = __builtin_shufflevector(alo, ahi, 0,1,2,3,4,5,6,7,8,9,10,11,12,13,14,15);
    v16bf b0 = *(const v16bf*)(Bw + (long)(g0 +  0 + ln) * EE + k0 + 16 * half);
    v16bf b1 = *(const v16bf*)(Bw + (long)(g0 + 16 + ln) * EE + k0 + 16 * half);
    v16bf b2 = *(const v16bf*)(Bw + (long)(g0 + 32 + ln) * EE + k0 + 16 * half);
    v16bf b3 = *(const v16bf*)(Bw + (long)(g0 + 48 + ln) * EE + k0 + 16 * half);
    acc0 = __builtin_amdgcn_wmma_f32_16x16x32_bf16(false, a, false, b0, (short)0, acc0, false, false);
    acc1 = __builtin_amdgcn_wmma_f32_16x16x32_bf16(false, a, false, b1, (short)0, acc1, false, false);
    acc2 = __builtin_amdgcn_wmma_f32_16x16x32_bf16(false, a, false, b2, (short)0, acc2, false, false);
    acc3 = __builtin_amdgcn_wmma_f32_16x16x32_bf16(false, a, false, b3, (short)0, acc3, false, false);
  }

  v8f* accs[4] = {&acc0, &acc1, &acc2, &acc3};
  #pragma unroll
  for (int nt = 0; nt < 4; ++nt) {
    int g = g0 + nt * 16 + ln;
    #pragma unroll
    for (int r = 0; r < 8; ++r) {
      int m = m0 + r + 8 * half;
      int b = m & 31;
      Gp[(long)m * G4 + g] = (*accs[nt])[r] + constv[b * G4 + g];
    }
  }
}

// ---------------------------------------------------------------------------
// Kernel 5: persistent recurrence. 32 WGs x 256 threads; WG owns 16 h-cols.
//   Per step: async-DMA h_{s-1} (bf16, 32KB) global->LDS (ASYNCcnt path),
//   prefetch next step's Gp slice, 8 waves compute the 8 16x16 gate tiles
//   via WMMA (two independent K-chains), exchange via LDS, apply LSTM
//   elementwise, publish h_s (f32 to d_out, bf16 to double buffer), grid
//   barrier.
// ---------------------------------------------------------------------------
__global__ void __launch_bounds__(256) k_recurrent(const float* __restrict__ Gp,
                                                   const __bf16* __restrict__ Wh,
                                                   const float* __restrict__ c0,
                                                   float* __restrict__ out,
                                                   __bf16* __restrict__ hbf,
                                                   unsigned* __restrict__ bar) {
  __shared__ __bf16 hprev[HB * HH];    // 32 KB: h_{s-1} in bf16
  __shared__ float  gbuf[4][HB][16];   // 8 KB: gate tiles for this WG's cols

  const int tid  = threadIdx.x;
  const int lane = tid & 31;
  const int wave = tid >> 5;
  const int ln   = lane & 15;
  const int half = lane >> 4;
  const int wg   = blockIdx.x;         // 0..31
  const int hc0  = wg * 16;            // h-column base
  const int q    = wave >> 1;          // gate index 0..3 (i,f,g,o)
  const int m0   = (wave & 1) * 16;    // m tile base

  // LDS byte address of hprev (flat aperture: low 32 bits = LDS offset)
  const unsigned lds_base = (unsigned)(size_t)(&hprev[0]);
  const __bf16* brow = Wh + (long)(q * HH + hc0 + ln) * HH;  // B rows, L2-hot

  for (int s = 0; s < SS; ++s) {
    // ---- stage h_{s-1} into LDS (async DMA for s>0, zeros for s==0) ----
    if (s == 0) {
      for (int i = tid; i < HB * HH; i += 256) hprev[i] = (__bf16)0.0f;
    } else {
      unsigned long long src =
          (unsigned long long)(size_t)(hbf + ((s - 1) & 1) * (HB * HH));
      #pragma unroll
      for (int it = 0; it < (HB * HH * 2) / (16 * 256); ++it) {  // 8 iters
        int i = tid + it * 256;                  // 16-byte chunk index
        unsigned laddr = lds_base + (unsigned)i * 16u;
        unsigned long long gaddr = src + (unsigned long long)i * 16ull;
        asm volatile("global_load_async_to_lds_b128 %0, %1, off"
                     :: "v"(laddr), "v"(gaddr) : "memory");
      }
      asm volatile("s_wait_asynccnt 0x0" ::: "memory");
    }
    __syncthreads();

    // ---- prefetch next step's Gp slice into cache while we compute ----
    if (s + 1 < SS) {
      const float* gpn = Gp + (long)(s + 1) * (HB * G4) + q * HH + hc0;
      #pragma unroll
      for (int r = 0; r < 8; ++r)
        __builtin_prefetch(gpn + (long)(m0 + r + 8 * half) * G4, 0, 0);
    }

    // ---- WMMA: 16x16 tile of (h_{s-1} @ Wh^T), two independent K-chains ----
    v8f accA = {}, accB = {};
    const __bf16* arow = &hprev[(m0 + ln) * HH];
    #pragma unroll
    for (int k0 = 0; k0 < HH / 2; k0 += 32) {
      int kA = k0, kB = k0 + HH / 2;
      v8bf aloA = *(const v8bf*)(arow + kA + 16 * half);
      v8bf ahiA = *(const v8bf*)(arow + kA + 8 + 16 * half);
      v16bf aA = __builtin_shufflevector(aloA, ahiA, 0,1,2,3,4,5,6,7,8,9,10,11,12,13,14,15);
      v16bf bA = *(const v16bf*)(brow + kA + 16 * half);
      accA = __builtin_amdgcn_wmma_f32_16x16x32_bf16(false, aA, false, bA, (short)0, accA, false, false);

      v8bf aloB = *(const v8bf*)(arow + kB + 16 * half);
      v8bf ahiB = *(const v8bf*)(arow + kB + 8 + 16 * half);
      v16bf aB = __builtin_shufflevector(aloB, ahiB, 0,1,2,3,4,5,6,7,8,9,10,11,12,13,14,15);
      v16bf bB = *(const v16bf*)(brow + kB + 16 * half);
      accB = __builtin_amdgcn_wmma_f32_16x16x32_bf16(false, aB, false, bB, (short)0, accB, false, false);
    }
    v8f acc = accA + accB;

    // ---- add precomputed emb_gates+const, publish tile to LDS ----
    const float* gp = Gp + (long)s * (HB * G4);
    #pragma unroll
    for (int r = 0; r < 8; ++r) {
      int m = m0 + r + 8 * half;
      gbuf[q][m][ln] = acc[r] + gp[(long)m * G4 + q * HH + hc0 + ln];
    }
    __syncthreads();

    // ---- elementwise LSTM cell: 32 x 16 values, 2 per thread ----
    __bf16* hb = hbf + (s & 1) * (HB * HH);
    for (int i = tid; i < HB * 16; i += 256) {
      int b = i >> 4, n = i & 15;
      float iv = gbuf[0][b][n], fv = gbuf[1][b][n];
      float gv = gbuf[2][b][n], ov = gbuf[3][b][n];
      float ig = 1.0f / (1.0f + __expf(-iv));
      float fg = 1.0f / (1.0f + __expf(-fv));
      float og = 1.0f / (1.0f + __expf(-ov));
      float gt = tanhf(gv);
      float cn = fg * c0[b * HH + hc0 + n] + ig * gt;
      float hn = og * tanhf(cn);
      out[(long)b * (SS * HH) + (long)s * HH + hc0 + n] = hn;
      hb[b * HH + hc0 + n] = (__bf16)hn;
    }

    // ---- grid barrier (release h_s, acquire before next step) ----
    __threadfence();
    __syncthreads();
    if (tid == 0) {
      __hip_atomic_fetch_add(bar, 1u, __ATOMIC_ACQ_REL, __HIP_MEMORY_SCOPE_AGENT);
      unsigned target = (unsigned)RWG * (unsigned)(s + 1);
      while (__hip_atomic_load(bar, __ATOMIC_ACQUIRE, __HIP_MEMORY_SCOPE_AGENT) < target) {
        __builtin_amdgcn_s_sleep(2);
      }
    }
    __syncthreads();
  }
}

// ---------------------------------------------------------------------------
// Host-side orchestration (graph-capture safe: only async stream ops).
// ---------------------------------------------------------------------------
extern "C" void kernel_launch(void* const* d_in, const int* in_sizes, int n_in,
                              void* d_out, int out_size, void* d_ws, size_t ws_size,
                              hipStream_t stream) {
  (void)in_sizes; (void)n_in; (void)out_size; (void)ws_size;

  const int*   seq   = (const int*)  d_in[0];
  // d_in[1] (enc_out) is unused by the reference
  const float* enc_h = (const float*)d_in[2];   // (1,B,H)
  const float* enc_c = (const float*)d_in[3];   // (1,B,H)
  const float* emb   = (const float*)d_in[4];   // (VOCAB,E)
  const float* W_ih  = (const float*)d_in[5];   // (4H, E+H)
  const float* W_hh  = (const float*)d_in[6];   // (4H, H)
  const float* b_ih  = (const float*)d_in[7];
  const float* b_hh  = (const float*)d_in[8];
  float* out = (float*)d_out;

  // Workspace layout (256B aligned blocks)
  char* ws = (char*)d_ws;
  unsigned* bar    = (unsigned*)ws;        size_t off = 256;
  __bf16*   We     = (__bf16*)(ws + off);  off += (size_t)G4 * EE * 2;       // 2 MB
  __bf16*   Wh     = (__bf16*)(ws + off);  off += (size_t)G4 * HH * 2;       // 2 MB
  __bf16*   Aemb   = (__bf16*)(ws + off);  off += (size_t)SS * HB * EE * 2;  // 16 MB
  float*    constv = (float*)(ws + off);   off += (size_t)HB * G4 * 4;       // 256 KB
  __bf16*   hbf    = (__bf16*)(ws + off);  off += (size_t)2 * HB * HH * 2;   // 64 KB
  float*    Gp     = (float*)(ws + off);   // 512*32*2048*4 = 134 MB

  hipMemsetAsync(bar, 0, sizeof(unsigned), stream);

  k_convert_w<<<(G4 * (EE + HH) / 8) / 256, 256, 0, stream>>>(W_ih, We, Wh);
  k_embed    <<<(SS * HB * EE / 8) / 256,   256, 0, stream>>>(seq, emb, Aemb);
  k_const    <<<(HB * G4) / 256,            256, 0, stream>>>(enc_h, W_hh, b_ih, b_hh, constv);
  k_gemm_emb <<<dim3(G4 / 64, (SS * HB / 16) / 8), 256, 0, stream>>>(Aemb, We, constv, Gp);
  k_recurrent<<<RWG, 256, 0, stream>>>(Gp, Wh, enc_c, out, hbf, bar);
}
